// GraphSAGE_54185307406416
// MI455X (gfx1250) — matrix operand order repaired and checked by reading, not compiled, not run on hardware
//
#include <hip/hip_runtime.h>

// ---------------------------------------------------------------------------
// GraphSAGE (2-layer, mean aggregation) for MI455X / gfx1250, wave32 + WMMA.
//   h1 = relu(x @ Ws1 + mean_agg(x) @ Wn1 + b1)
//   y  =      h1 @ Ws2 + mean_agg(h1) @ Wn2 + b2
// GEMMs run on v_wmma_f32_16x16x32_f16 (f16 inputs, f32 accumulate).
// Aggregation = f32 atomic scatter-add (L2-resident: tables fit in 192MB L2).
// ---------------------------------------------------------------------------

typedef __attribute__((ext_vector_type(16))) _Float16 v16h;
typedef __attribute__((ext_vector_type(8)))  _Float16 v8h;
typedef __attribute__((ext_vector_type(4)))  _Float16 v4h;
typedef __attribute__((ext_vector_type(8)))  float    v8f;

#define NODES 100000
#define EDGES 640000
#define K128  128

// ---- zero a float4 region -------------------------------------------------
__global__ void zero_f4(float4* __restrict__ p, int n4) {
  int i = blockIdx.x * blockDim.x + threadIdx.x;
  if (i < n4) p[i] = make_float4(0.f, 0.f, 0.f, 0.f);
}

// ---- f32 -> f16 convert, 4 elements/thread --------------------------------
__global__ void cvt_f32_to_f16(const float4* __restrict__ s, v4h* __restrict__ d, int n4) {
  int i = blockIdx.x * blockDim.x + threadIdx.x;
  if (i >= n4) return;
  float4 v = s[i];
  v4h h;
  h[0] = (_Float16)v.x; h[1] = (_Float16)v.y;
  h[2] = (_Float16)v.z; h[3] = (_Float16)v.w;
  d[i] = h;
}

// ---- pack a KxN (K=128, row-major, f32) weight into per-lane WMMA B
//      fragments: Wp[((ntile*4 + kstep)*32 + lane)*16 + e]
//        = (f16) W[(kstep*32 + (lane<16?0:8) + (e&7) + (e>>3)*16)*N + ntile*16 + (lane&15)]
__global__ void pack_w(const float* __restrict__ W, _Float16* __restrict__ Wp, int N) {
  int idx = blockIdx.x * blockDim.x + threadIdx.x;
  if (idx >= 128 * N) return;
  int e     = idx & 15;
  int lane  = (idx >> 4) & 31;
  int kstep = (idx >> 9) & 3;
  int ntile = idx >> 11;
  int k = kstep * 32 + ((lane < 16) ? 0 : 8) + (e & 7) + ((e >> 3) << 4);
  int n = ntile * 16 + (lane & 15);
  Wp[idx] = (_Float16)W[k * N + n];
}

// ---- in-degree (same for both layers) -------------------------------------
__global__ void degree_k(const int* __restrict__ dst, float* __restrict__ deg) {
  int e = blockIdx.x * blockDim.x + threadIdx.x;
  if (e < EDGES) atomicAdd(&deg[dst[e]], 1.0f);
}

// ---- scatter-add: one wave per edge, 4 feats/lane (128-wide rows) ---------
__global__ void aggregate_k(const _Float16* __restrict__ feats,
                            const int* __restrict__ src, const int* __restrict__ dst,
                            float* __restrict__ agg) {
  int wave = blockIdx.x * 8 + (threadIdx.x >> 5);
  int lane = threadIdx.x & 31;
  if (wave >= EDGES) return;
  int s = src[wave], d = dst[wave];
  v4h h = *(const v4h*)(feats + (size_t)s * K128 + lane * 4);
  float* a = agg + (size_t)d * K128 + lane * 4;
  atomicAdd(a + 0, (float)h[0]);
  atomicAdd(a + 1, (float)h[1]);
  atomicAdd(a + 2, (float)h[2]);
  atomicAdd(a + 3, (float)h[3]);
}

// ---- mean (zero-degree safe) + convert to f16 + re-zero agg for next layer
__global__ void mean_cvt_zero(const float* __restrict__ deg,
                              float4* __restrict__ agg4, v4h* __restrict__ nh4, int n4) {
  int i = blockIdx.x * blockDim.x + threadIdx.x;
  if (i >= n4) return;
  int node = i >> 5;                       // 32 float4 per 128-wide row
  float inv = 1.0f / fmaxf(deg[node], 1.0f);
  float4 a = agg4[i];
  v4h h;
  h[0] = (_Float16)(a.x * inv); h[1] = (_Float16)(a.y * inv);
  h[2] = (_Float16)(a.z * inv); h[3] = (_Float16)(a.w * inv);
  nh4[i] = h;
  agg4[i] = make_float4(0.f, 0.f, 0.f, 0.f);
}

// ---- fused dual-GEMM: out = A1@W1 + A2@W2 + bias (K=128, one 16x16 tile/wave)
template <bool RELU, bool OUT32>
__global__ void sage_gemm(const _Float16* __restrict__ A1, const _Float16* __restrict__ A2,
                          const _Float16* __restrict__ W1p, const _Float16* __restrict__ W2p,
                          const float* __restrict__ bias,
                          float* __restrict__ out32, _Float16* __restrict__ out16,
                          int Ntiles, int N) {
  int wave = blockIdx.x * 8 + (threadIdx.x >> 5);   // grid sized exactly
  int lane = threadIdx.x & 31;
  int tm = wave / Ntiles;
  int tn = wave - tm * Ntiles;
  int l15    = lane & 15;
  int hiHalf = lane >> 4;                           // 0 | 1
  int row = tm * 16 + l15;
  int col = tn * 16 + l15;

  const _Float16* a1p = A1 + (size_t)row * K128 + hiHalf * 8;   // ISA A layout
  const _Float16* a2p = A2 + (size_t)row * K128 + hiHalf * 8;
  const v16h* w1 = (const v16h*)(W1p + ((size_t)(tn * 4) * 32 + lane) * 16);
  const v16h* w2 = (const v16h*)(W2p + ((size_t)(tn * 4) * 32 + lane) * 16);

  float bv = bias[col];
  v8f acc = {bv, bv, bv, bv, bv, bv, bv, bv};

#pragma unroll
  for (int ks = 0; ks < 4; ++ks) {                  // K = 4 * 32
    v8h lo1 = *(const v8h*)(a1p + ks * 32);
    v8h hi1 = *(const v8h*)(a1p + ks * 32 + 16);
    v16h af1 = __builtin_shufflevector(lo1, hi1, 0,1,2,3,4,5,6,7,8,9,10,11,12,13,14,15);
    v16h bf1 = w1[ks * 32];
    acc = __builtin_amdgcn_wmma_f32_16x16x32_f16(false, af1, false, bf1,
                                                 (short)0, acc, false, false);
    v8h lo2 = *(const v8h*)(a2p + ks * 32);
    v8h hi2 = *(const v8h*)(a2p + ks * 32 + 16);
    v16h af2 = __builtin_shufflevector(lo2, hi2, 0,1,2,3,4,5,6,7,8,9,10,11,12,13,14,15);
    v16h bf2 = w2[ks * 32];
    acc = __builtin_amdgcn_wmma_f32_16x16x32_f16(false, af2, false, bf2,
                                                 (short)0, acc, false, false);
  }

  int rbase = tm * 16 + hiHalf * 8;                 // ISA C/D layout
#pragma unroll
  for (int r = 0; r < 8; ++r) {
    float v = acc[r];
    if (RELU) v = fmaxf(v, 0.f);
    size_t o = (size_t)(rbase + r) * N + col;
    if (OUT32) out32[o] = v;
    else       out16[o] = (_Float16)v;
  }
}

// ---------------------------------------------------------------------------
extern "C" void kernel_launch(void* const* d_in, const int* in_sizes, int n_in,
                              void* d_out, int out_size, void* d_ws, size_t ws_size,
                              hipStream_t stream) {
  (void)in_sizes; (void)n_in; (void)out_size; (void)ws_size;
  const float* x   = (const float*)d_in[0];
  const int*   es  = (const int*)d_in[1];
  const int*   ed  = (const int*)d_in[2];
  const float* Ws1 = (const float*)d_in[3];
  const float* Wn1 = (const float*)d_in[4];
  const float* b1  = (const float*)d_in[5];
  const float* Ws2 = (const float*)d_in[6];
  const float* Wn2 = (const float*)d_in[7];
  const float* b2  = (const float*)d_in[8];

  // workspace layout (~128.5 MB), all offsets 16B-aligned
  char* ws = (char*)d_ws;
  float*    deg = (float*)ws;                       //    400,000 B
  float*    agg = (float*)(ws + 400000);            // 51,200,000 B
  _Float16* xh  = (_Float16*)(ws + 51600000);       // 25,600,000 B
  _Float16* nh  = (_Float16*)(ws + 77200000);       // 25,600,000 B
  _Float16* hh  = (_Float16*)(ws + 102800000);      // 25,600,000 B
  _Float16* wp  = (_Float16*)(ws + 128400000);      //     98,304 B packed weights
  _Float16* wp_s1 = wp;                             // 128x128
  _Float16* wp_n1 = wp + 16384;                     // 128x128
  _Float16* wp_s2 = wp + 32768;                     // 128x64
  _Float16* wp_n2 = wp + 40960;                     // 128x64

  const int nF4 = NODES * K128 / 4;                 // 3,200,000 float4 per table
  const int nZ4 = (NODES + NODES * K128) / 4;       // deg + agg, contiguous

  // deg + agg := 0
  zero_f4<<<(nZ4 + 255) / 256, 256, 0, stream>>>((float4*)ws, nZ4);
  // x -> f16
  cvt_f32_to_f16<<<(nF4 + 255) / 256, 256, 0, stream>>>((const float4*)x, (v4h*)xh, nF4);
  // pack weights into WMMA B-fragment layout
  pack_w<<<(128 * 128 + 255) / 256, 256, 0, stream>>>(Ws1, wp_s1, 128);
  pack_w<<<(128 * 128 + 255) / 256, 256, 0, stream>>>(Wn1, wp_n1, 128);
  pack_w<<<(128 *  64 + 255) / 256, 256, 0, stream>>>(Ws2, wp_s2, 64);
  pack_w<<<(128 *  64 + 255) / 256, 256, 0, stream>>>(Wn2, wp_n2, 64);
  // degrees (shared by both layers)
  degree_k<<<EDGES / 256, 256, 0, stream>>>(ed, deg);

  // ---- layer 1 ----
  aggregate_k<<<EDGES / 8, 256, 0, stream>>>(xh, es, ed, agg);
  mean_cvt_zero<<<(nF4 + 255) / 256, 256, 0, stream>>>(deg, (float4*)agg, (v4h*)nh, nF4);
  // h = relu(x@Ws1 + nh@Wn1 + b1): 6250x8 = 50000 tiles, 8 waves/block
  sage_gemm<true, false><<<6250, 256, 0, stream>>>(xh, nh, wp_s1, wp_n1, b1,
                                                   nullptr, hh, 8, 128);

  // ---- layer 2 ----
  aggregate_k<<<EDGES / 8, 256, 0, stream>>>(hh, es, ed, agg);
  mean_cvt_zero<<<(nF4 + 255) / 256, 256, 0, stream>>>(deg, (float4*)agg, (v4h*)nh, nF4);
  // out = h@Ws2 + nh@Wn2 + b2: 6250x4 = 25000 tiles
  sage_gemm<false, true><<<3125, 256, 0, stream>>>(hh, nh, wp_s2, wp_n2, b2,
                                                   (float*)d_out, nullptr, 4, 64);
}
